// EX_Module_7670811590899
// MI455X (gfx1250) — matrix operand order repaired and checked by reading, not compiled, not run on hardware
//
#include <hip/hip_runtime.h>
#include <cstdint>
#include <cstddef>

#define NB 16
#define NC 512
#define NHW 4096

typedef __attribute__((ext_vector_type(16))) _Float16 v16h;
typedef __attribute__((ext_vector_type(8)))  float    v8f;

__device__ __forceinline__ float hsig(float v){ v=(v+3.0f)*(1.0f/6.0f); return v<0.f?0.f:(v>1.f?1.f:v); }
__device__ __forceinline__ float sigm(float v){ return 1.0f/(1.0f+__expf(-v)); }

template<int NT> __device__ float block_sum(float v){
  __shared__ float s[NT];
  s[threadIdx.x]=v; __syncthreads();
  for(int o=NT/2;o>0;o>>=1){ if((int)threadIdx.x<o) s[threadIdx.x]+=s[threadIdx.x+o]; __syncthreads(); }
  float r=s[0]; __syncthreads(); return r;
}
template<int NT> __device__ float block_max(float v){
  __shared__ float s[NT];
  s[threadIdx.x]=v; __syncthreads();
  for(int o=NT/2;o>0;o>>=1){ if((int)threadIdx.x<o) s[threadIdx.x]=fmaxf(s[threadIdx.x],s[threadIdx.x+o]); __syncthreads(); }
  float r=s[0]; __syncthreads(); return r;
}

// LDS byte offset of a shared-memory pointer (LDS aperture occupies addr[63:32];
// low 32 bits are the in-LDS offset per ISA 10.2 aperture mapping)
__device__ __forceinline__ uint32_t lds_off(const void* p){ return (uint32_t)(uintptr_t)p; }

// CDNA5 async copy: 16B from per-lane global address to per-lane LDS address (ASYNCcnt)
__device__ __forceinline__ void async_load_b128(uint32_t ldsaddr, const void* gptr){
  asm volatile("global_load_async_to_lds_b128 %0, %1, off"
               :: "v"(ldsaddr), "v"(gptr) : "memory");
}
__device__ __forceinline__ void wait_async_le6(){ asm volatile("s_wait_asynccnt 0x6" ::: "memory"); }
__device__ __forceinline__ void wait_async_0 (){ asm volatile("s_wait_asynccnt 0x0" ::: "memory"); }

// ---- x (b,c,hw) f32 -> xt (b,hw,c) f16, 32x32 LDS tiles ----
__global__ __launch_bounds__(256) void k_transpose_cvt(const float* __restrict__ x,
                                                       _Float16* __restrict__ xt){
  __shared__ float tile[32][33];
  const int b = blockIdx.z, c0 = blockIdx.y*32, n0 = blockIdx.x*32;
  const int tx = threadIdx.x, ty = threadIdx.y;     // (32,8)
  const float* xp = x + ((size_t)b*NC + c0)*NHW + n0;
  #pragma unroll
  for(int i=0;i<4;++i) tile[ty+i*8][tx] = xp[(size_t)(ty+i*8)*NHW + tx];
  __syncthreads();
  _Float16* op = xt + ((size_t)b*NHW + n0)*NC + c0;
  #pragma unroll
  for(int i=0;i<4;++i) op[(size_t)(ty+i*8)*NC + tx] = (_Float16)tile[tx][ty+i*8];
}

__global__ void k_f32_to_f16(const float* __restrict__ in, _Float16* __restrict__ out, int n){
  int i = blockIdx.x*blockDim.x + threadIdx.x;
  if(i < n) out[i] = (_Float16)in[i];
}

// ---- WMMA GEMM: Y[b,m,n] = relu(sum_c W[m,c]*X[b,c,n]) ----
// Block = 4 waves, tile M=64 x N=128, K-step 32.
// Triple-buffered LDS fed by global_load_async_to_lds_b128; one barrier per K-step.
// REDUCE=0: store Y f16 (b,c,hw).  REDUCE=1: write per-(b,m) partial HW-sums.
template<int REDUCE>
__global__ __launch_bounds__(128) void k_conv_gemm(const _Float16* __restrict__ W,
                                                   const _Float16* __restrict__ Xt,
                                                   _Float16* __restrict__ Y,
                                                   float* __restrict__ partial){
  __shared__ _Float16 Ws[3][64*32];    // 3 x 4KB : W tile, row-major 32 halves/row
  __shared__ _Float16 Xs[3][128*32];   // 3 x 8KB : X tile, row(=n)-major 32 halves/row

  const int t    = threadIdx.x;
  const int lane = t & 31;
  const int wave = t >> 5;
  const int lrow = lane & 15;
  const int lhi  = lane >> 4;
  const int b  = blockIdx.z;
  const int m0blk = blockIdx.y*64;
  const int n0    = blockIdx.x*128;

  const _Float16* Wblk = W  + (size_t)m0blk*NC;
  const _Float16* Xblk = Xt + ((size_t)b*NHW + n0)*NC;

  uint32_t wsb[3] = { lds_off(&Ws[0][0]), lds_off(&Ws[1][0]), lds_off(&Ws[2][0]) };
  uint32_t xsb[3] = { lds_off(&Xs[0][0]), lds_off(&Xs[1][0]), lds_off(&Xs[2][0]) };

  // stage one K-step tile (W: 256 chunks of 16B, X: 512 chunks) into LDS buffer `buf`
  auto stage = [&](int buf, int k){
    #pragma unroll
    for(int i=0;i<2;++i){
      const int cid = t + i*128;                 // 0..255
      const int row = cid >> 2, c4 = cid & 3;
      async_load_b128(wsb[buf] + cid*16, (const char*)(Wblk + (size_t)row*NC + k) + c4*16);
    }
    #pragma unroll
    for(int i=0;i<4;++i){
      const int cid = t + i*128;                 // 0..511
      const int row = cid >> 2, c4 = cid & 3;
      async_load_b128(xsb[buf] + cid*16, (const char*)(Xblk + (size_t)row*NC + k) + c4*16);
    }
  };

  const v8f vzero = {0.f,0.f,0.f,0.f,0.f,0.f,0.f,0.f};
  v8f acc[8];
  #pragma unroll
  for(int s=0;s<8;++s) acc[s] = vzero;

  stage(0, 0);
  stage(1, 32);
  #pragma unroll
  for(int i=0;i<16;++i){
    const int cur = i % 3;
    if(i < 15) wait_async_le6(); else wait_async_0();   // stage i complete (in-order)
    __syncthreads();             // all waves: stage i visible AND compute i-1 finished
    if(i+2 < 16) stage((i+2) % 3, (i+2)*32);            // buffer last read at i-1: free

    // A fragment (16x32): lane halves hold K 0..15 / 16..31
    v16h a;
    const _Float16* wr = &Ws[cur][(wave*16 + lrow)*32 + lhi*16];
    #pragma unroll
    for(int j=0;j<8;++j){ a[j] = wr[j]; a[j+8] = wr[8+j]; }

    #pragma unroll
    for(int s=0;s<8;++s){
      // B fragment (32x16): lane = n, halves split K 0..15 / 16..31; contiguous 32B in LDS
      v16h bf = *(const v16h*)&Xs[cur][(s*16 + lrow)*32 + lhi*16];
      acc[s] = __builtin_amdgcn_wmma_f32_16x16x32_f16(false, a, false, bf,
                                                      (short)0, acc[s], false, false);
    }
  }

  const int m0 = m0blk + wave*16;
  if(!REDUCE){
    _Float16* Yb = Y + (size_t)b*NC*NHW;
    #pragma unroll
    for(int r=0;r<8;++r){
      const int m = m0 + r + lhi*8;
      #pragma unroll
      for(int s=0;s<8;++s){
        float v = acc[s][r]; v = v>0.f ? v : 0.f;
        Yb[(size_t)m*NHW + n0 + s*16 + lrow] = (_Float16)v;
      }
    }
  } else {
    #pragma unroll
    for(int r=0;r<8;++r){
      float v = 0.f;
      #pragma unroll
      for(int s=0;s<8;++s){ float q = acc[s][r]; v += (q>0.f ? q : 0.f); }
      for(int o=1;o<16;o<<=1) v += __shfl_xor(v, o, 16);
      if(lrow==0){
        const int m = m0 + r + lhi*8;
        partial[((size_t)b*NC + m)*32 + blockIdx.x] = v;
      }
    }
  }
}

// ---- sum 32 partials -> gsum[b,c] ----
__global__ __launch_bounds__(32) void k_reduce_partial(const float* __restrict__ partial,
                                                       float* __restrict__ gsum){
  const int b = blockIdx.y, c = blockIdx.x;
  float v = partial[((size_t)b*NC + c)*32 + threadIdx.x];
  v = block_sum<32>(v);
  if(threadIdx.x==0) gsum[b*NC + c] = v;
}

// ---- mask[b,n] = relu(sum_c wq[c]*x[b,c,n]) ----
__global__ __launch_bounds__(256) void k_mask(const float* __restrict__ x,
                                              const float* __restrict__ wq,
                                              float* __restrict__ mask){
  __shared__ float w[NC];
  const int b = blockIdx.y, n = blockIdx.x*256 + threadIdx.x;
  for(int c=threadIdx.x;c<NC;c+=256) w[c] = wq[c];
  __syncthreads();
  const float* xp = x + (size_t)b*NC*NHW + n;
  float acc = 0.f;
  for(int c=0;c<NC;++c) acc += w[c]*xp[(size_t)c*NHW];
  mask[b*NHW + n] = fmaxf(acc, 0.f);
}

// ---- softmax over hw (4096) per batch ----
__global__ __launch_bounds__(256) void k_softmax_hw(const float* __restrict__ in,
                                                    float* __restrict__ out){
  const int b = blockIdx.x;
  const float* ip = in + (size_t)b*NHW;
  float m = -3.4e38f;
  for(int i=threadIdx.x;i<NHW;i+=256) m = fmaxf(m, ip[i]);
  m = block_max<256>(m);
  float s = 0.f;
  for(int i=threadIdx.x;i<NHW;i+=256) s += __expf(ip[i]-m);
  s = block_sum<256>(s);
  float inv = 1.0f/s;
  float* op = out + (size_t)b*NHW;
  for(int i=threadIdx.x;i<NHW;i+=256) op[i] = __expf(ip[i]-m)*inv;
}

// ---- context[b,c] = sum_n input_x[b,c,n]*p[b,n] ----
__global__ __launch_bounds__(128) void k_context(const _Float16* __restrict__ inx,
                                                 const float* __restrict__ p,
                                                 float* __restrict__ context){
  const int c = blockIdx.x, b = blockIdx.y;
  const _Float16* ip = inx + ((size_t)b*NC + c)*NHW;
  const float* pp = p + (size_t)b*NHW;
  float acc = 0.f;
  for(int n=threadIdx.x;n<NHW;n+=128) acc += (float)ip[n]*pp[n];
  acc = block_sum<128>(acc);
  if(threadIdx.x==0) context[b*NC + c] = acc;
}

// ---- LN(c) -> sigmoid -> w_up GEMV -> hsigmoid ----
__global__ __launch_bounds__(512) void k_spatial(const float* __restrict__ context,
                                                 const float* __restrict__ w_up,
                                                 float* __restrict__ spatial){
  __shared__ float sh[NC];
  const int b = blockIdx.x, c = threadIdx.x;
  float v = context[b*NC + c];
  float mu = block_sum<512>(v)*(1.0f/NC);
  float d = v - mu;
  float var = block_sum<512>(d*d)*(1.0f/NC);
  float ln = d*rsqrtf(var + 1e-5f);
  sh[c] = sigm(ln);
  __syncthreads();
  const float* wr = w_up + (size_t)c*NC;
  float acc = 0.f;
  for(int j=0;j<NC;++j) acc += wr[j]*sh[j];
  spatial[b*NC + c] = hsig(acc);
}

// ---- avg_x[b,c] = softmax_c(gsum/HW) ----
__global__ __launch_bounds__(512) void k_avg(const float* __restrict__ gsum,
                                             float* __restrict__ avg){
  const int b = blockIdx.x, c = threadIdx.x;
  float v = gsum[b*NC + c]*(1.0f/NHW);
  float m = block_max<512>(v);
  float e = __expf(v - m);
  float s = block_sum<512>(e);
  avg[b*NC + c] = e/s;
}

// ---- ch[b,n] = sigmoid(sum_c avg[b,c]*theta[b,c,n]) ----
__global__ __launch_bounds__(256) void k_channel(const _Float16* __restrict__ theta,
                                                 const float* __restrict__ avg,
                                                 float* __restrict__ ch){
  __shared__ float av[NC];
  const int b = blockIdx.y, n = blockIdx.x*256 + threadIdx.x;
  for(int c=threadIdx.x;c<NC;c+=256) av[c] = avg[b*NC + c];
  __syncthreads();
  const _Float16* tp = theta + (size_t)b*NC*NHW + n;
  float acc = 0.f;
  for(int c=0;c<NC;++c) acc += av[c]*(float)tp[(size_t)c*NHW];
  ch[b*NHW + n] = sigm(acc);
}

// ---- per-batch ch stats: mean + max ----
__global__ __launch_bounds__(256) void k_chstats(const float* __restrict__ ch,
                                                 float* __restrict__ mch,
                                                 float* __restrict__ maxch){
  const int b = blockIdx.x;
  const float* cp = ch + (size_t)b*NHW;
  float s = 0.f, m = -3.4e38f;
  for(int i=threadIdx.x;i<NHW;i+=256){ float v = cp[i]; s += v; m = fmaxf(m, v); }
  s = block_sum<256>(s);
  m = block_max<256>(m);
  if(threadIdx.x==0){ mch[b] = s*(1.0f/NHW); maxch[b] = m; }
}

// ---- SK gating: u -> w_sk1 -> relu -> w_sk2 -> relu -> softmax_c -> Acoef ----
__global__ __launch_bounds__(512) void k_sk(const float* __restrict__ spatial,
                                            const float* __restrict__ mch,
                                            const float* __restrict__ w_sk1,
                                            const float* __restrict__ w_sk2,
                                            float* __restrict__ Acoef){
  __shared__ float u[NC];
  __shared__ float u1[NC/2];
  const int b = blockIdx.x, c = threadIdx.x;
  const float s = spatial[b*NC + c];
  const float m = mch[b];
  u[c] = s*(1.0f + m) + m;               // mean_n(seq+par) = s*mch + s + mch
  __syncthreads();
  if(c < NC/2){
    const float* wr = w_sk1 + (size_t)c*NC;
    float acc = 0.f;
    for(int j=0;j<NC;++j) acc += wr[j]*u[j];
    u1[c] = fmaxf(acc, 0.f);
  }
  __syncthreads();
  const float* wr2 = w_sk2 + (size_t)c*(NC/2);
  float tv = 0.f;
  for(int j=0;j<NC/2;++j) tv += wr2[j]*u1[j];
  tv = fmaxf(tv, 0.f);
  float mx = block_max<512>(tv);
  float e = __expf(tv - mx);
  float z = block_sum<512>(e);
  float a = e/z;
  // sk = a*(s*ch) + (1-a)*(s+ch) = A*ch + B,  A = a*s+1-a (>=0), B cancels in softmax_n
  Acoef[b*NC + c] = a*s + 1.0f - a;
}

// ---- final: softmax_n(A*ch) dot value, + residual x ----
__global__ __launch_bounds__(256) void k_out(const float* __restrict__ Acoef,
                                             const float* __restrict__ ch,
                                             const float* __restrict__ maxch,
                                             const _Float16* __restrict__ value,
                                             const float* __restrict__ x,
                                             float* __restrict__ out){
  __shared__ float chs[NHW];
  const int c = blockIdx.x, b = blockIdx.y;
  for(int i=threadIdx.x;i<NHW;i+=256) chs[i] = ch[(size_t)b*NHW + i];
  __syncthreads();
  const float A = Acoef[b*NC + c];
  const float M = maxch[b];
  const _Float16* vp = value + ((size_t)b*NC + c)*NHW;
  float z = 0.f, sv = 0.f;
  for(int n=threadIdx.x;n<NHW;n+=256){
    float e = __expf(A*(chs[n] - M));
    z += e; sv += e*(float)vp[n];
  }
  z = block_sum<256>(z);
  sv = block_sum<256>(sv);
  const float r = sv/z;
  const float* xp = x + ((size_t)b*NC + c)*NHW;
  float* op = out + ((size_t)b*NC + c)*NHW;
  for(int n=threadIdx.x;n<NHW;n+=256) op[n] = r + xp[n];
}

extern "C" void kernel_launch(void* const* d_in, const int* in_sizes, int n_in,
                              void* d_out, int out_size, void* d_ws, size_t ws_size,
                              hipStream_t stream){
  const float* x         = (const float*)d_in[0];
  const float* w_q_right = (const float*)d_in[1];
  const float* w_v_right = (const float*)d_in[2];
  const float* w_up      = (const float*)d_in[3];
  const float* w_q_left  = (const float*)d_in[4];
  const float* w_v_left  = (const float*)d_in[5];
  const float* w_sk1     = (const float*)d_in[6];
  const float* w_sk2     = (const float*)d_in[7];
  const float* w_res     = (const float*)d_in[8];
  float* out = (float*)d_out;

  char* p = (char*)d_ws;
  auto alloc = [&](size_t bytes)->void*{ void* r = (void*)p; p += (bytes + 255) & ~(size_t)255; return r; };
  const size_t NTOT = (size_t)NB*NC*NHW;

  _Float16* xt    = (_Float16*)alloc(NTOT*2);          // x transposed (b,hw,c)
  _Float16* h_wvr = (_Float16*)alloc((size_t)NC*NC*2);
  _Float16* h_wql = (_Float16*)alloc((size_t)NC*NC*2);
  _Float16* h_wvl = (_Float16*)alloc((size_t)NC*NC*2);
  _Float16* h_wrs = (_Float16*)alloc((size_t)NC*NC*2);
  _Float16* inx   = (_Float16*)alloc(NTOT*2);          // relu(w_v_right x)
  _Float16* theta = (_Float16*)alloc(NTOT*2);          // relu(w_v_left x)
  _Float16* value = (_Float16*)alloc(NTOT*2);          // relu(w_res x)
  float* partial  = (float*)alloc((size_t)NB*NC*32*4);
  float* gsum     = (float*)alloc((size_t)NB*NC*4);
  float* mask     = (float*)alloc((size_t)NB*NHW*4);
  float* psm      = (float*)alloc((size_t)NB*NHW*4);
  float* context  = (float*)alloc((size_t)NB*NC*4);
  float* spatial  = (float*)alloc((size_t)NB*NC*4);
  float* avg      = (float*)alloc((size_t)NB*NC*4);
  float* ch       = (float*)alloc((size_t)NB*NHW*4);
  float* mch      = (float*)alloc(NB*4);
  float* mxch     = (float*)alloc(NB*4);
  float* Acoef    = (float*)alloc((size_t)NB*NC*4);

  // 1) convert / transpose inputs to f16
  k_transpose_cvt<<<dim3(NHW/32, NC/32, NB), dim3(32,8), 0, stream>>>(x, xt);
  k_f32_to_f16<<<(NC*NC+255)/256, 256, 0, stream>>>(w_v_right, h_wvr, NC*NC);
  k_f32_to_f16<<<(NC*NC+255)/256, 256, 0, stream>>>(w_q_left,  h_wql, NC*NC);
  k_f32_to_f16<<<(NC*NC+255)/256, 256, 0, stream>>>(w_v_left,  h_wvl, NC*NC);
  k_f32_to_f16<<<(NC*NC+255)/256, 256, 0, stream>>>(w_res,     h_wrs, NC*NC);

  // 2) the four heavy GEMMs on WMMA (LDS triple-buffered async pipeline)
  dim3 ggrid(NHW/128, NC/64, NB);
  k_conv_gemm<0><<<ggrid, 128, 0, stream>>>(h_wvr, xt, inx,   nullptr);
  k_conv_gemm<0><<<ggrid, 128, 0, stream>>>(h_wvl, xt, theta, nullptr);
  k_conv_gemm<0><<<ggrid, 128, 0, stream>>>(h_wrs, xt, value, nullptr);
  k_conv_gemm<1><<<ggrid, 128, 0, stream>>>(h_wql, xt, nullptr, partial);
  k_reduce_partial<<<dim3(NC, NB), 32, 0, stream>>>(partial, gsum);

  // 3) right branch: mask softmax + context + spatial attention
  k_mask<<<dim3(NHW/256, NB), 256, 0, stream>>>(x, w_q_right, mask);
  k_softmax_hw<<<NB, 256, 0, stream>>>(mask, psm);
  k_context<<<dim3(NC, NB), 128, 0, stream>>>(inx, psm, context);
  k_spatial<<<NB, 512, 0, stream>>>(context, w_up, spatial);

  // 4) left branch: channel attention
  k_avg<<<NB, 512, 0, stream>>>(gsum, avg);
  k_channel<<<dim3(NHW/256, NB), 256, 0, stream>>>(theta, avg, ch);
  k_chstats<<<NB, 256, 0, stream>>>(ch, mch, mxch);

  // 5) SK gating + final fused softmax/dot/residual
  k_sk<<<NB, 512, 0, stream>>>(spatial, mch, w_sk1, w_sk2, Acoef);
  k_out<<<dim3(NC, NB), 256, 0, stream>>>(Acoef, ch, mxch, value, x, out);
}
